// VQBgModule_1099511628051
// MI455X (gfx1250) — matrix operand order repaired and checked by reading, not compiled, not run on hardware
//
#include <hip/hip_runtime.h>
#include <hip/hip_bf16.h>
#include <math.h>

typedef __attribute__((ext_vector_type(16))) _Float16 v16h;
typedef __attribute__((ext_vector_type(2)))  _Float16 v2h;
typedef __attribute__((ext_vector_type(8)))  float    v8f;

// ---------------------------------------------------------------------------
// Weight convert: f32 (Co, Kdim) row-major -> f16 (Co, Kpad) zero-padded
// ---------------------------------------------------------------------------
__global__ void wcvt_kernel(const float* __restrict__ w, _Float16* __restrict__ o,
                            int Co, int Kdim, int Kpad) {
    int i = blockIdx.x * 256 + threadIdx.x;
    if (i >= Co * Kpad) return;
    int co = i / Kpad;
    int k  = i - co * Kpad;
    o[i] = (k < Kdim) ? (_Float16)w[co * Kdim + k] : (_Float16)0.f;
}

// ---------------------------------------------------------------------------
// Implicit-GEMM convolution (FC = 1x1 conv with H=W=OH=OW=1).
// Block: 256 threads = 8 waves (wave32), tile M=64 x N=64, K-chunk 32.
// Waves arranged 4(M) x 2(N); each wave owns a 16x32 strip -> 2 accumulators,
// 2 x v_wmma_f32_16x16x32_f16 per K-step with the A fragment reused.
// A/B tiles are staged into LDS already in the ISA 7.12.2 per-lane fragment
// layout, so fragment reads are two aligned b128 LDS loads per matrix.
// Double-buffered LDS, one barrier per K-step. CELU/sigmoid + PixelShuffle(2)
// fused into the epilogue store.
// ---------------------------------------------------------------------------
__global__ __launch_bounds__(256) void conv_wmma_kernel(
        const float* __restrict__ x, const _Float16* __restrict__ wf,
        const float* __restrict__ bias, float* __restrict__ y,
        int M, int Co, int Ci, int H, int W,
        int KH, int S, int P, int OH, int OW,
        int Kdim, int Kpad, int act, int ps) {
    __shared__ alignas(16) _Float16 Al[2][4][32][16];   // [buf][m-sub][lane][h]
    __shared__ alignas(16) _Float16 Bl[2][4][32][16];   // [buf][n-sub][lane][h]

    const int tid    = threadIdx.x;
    const int lane   = tid & 31;
    const int wave   = tid >> 5;
    const int mw     = wave >> 1;      // 0..3 : wave's M sub-tile
    const int nw     = wave & 1;       // 0..1 : wave's N strip (2 sub-tiles)
    const int mblock = blockIdx.x * 64;
    const int nblock = blockIdx.y * 64;
    const int KK     = KH * KH;
    const int OHW    = OH * OW;

    // Per-thread A-gather coordinates (4 element-pairs, rows fixed across K)
    int an[4], aoh[4], aow[4];
#pragma unroll
    for (int i = 0; i < 4; ++i) {
        int pi  = tid + i * 256;       // pair index 0..1023
        int row = pi >> 4;             // 0..63
        int m   = mblock + row;
        if (m < M) {
            int n = m / OHW; int r = m - n * OHW;
            an[i] = n; aoh[i] = r / OW; aow[i] = r - (r / OW) * OW;
        } else an[i] = -1;
    }
    // Per-thread B-load coordinates (one 16B segment of one column)
    const int  bcol  = tid >> 2;           // 0..63
    const int  bseg  = tid & 3;            // 8-half segment within 32-K chunk
    const int  bcolg = nblock + bcol;
    const bool bval  = (bcolg < Co);
    const _Float16* bsrc = wf + (size_t)(bval ? bcolg : 0) * Kpad + bseg * 8;

    v8f acc0 = {}, acc1 = {};
    const int nstage = Kpad >> 5;
    for (int s = 0; s < nstage; ++s) {
        const int kc   = s << 5;
        const int pbuf = s & 1;

        // ---- stage A tile (scattered f32 gather -> f16 pairs in frag layout)
#pragma unroll
        for (int i = 0; i < 4; ++i) {
            int pi  = tid + i * 256;
            int row = pi >> 4;
            int kp  = (pi & 15) << 1;          // even k' within chunk
            float v0 = 0.f, v1 = 0.f;
            if (an[i] >= 0) {
                int k0 = kc + kp;
                if (k0 < Kdim) {
                    int ci = k0 / KK; int rem = k0 - ci * KK;
                    int kh = rem / KH; int kw = rem - kh * KH;
                    int ih = aoh[i] * S - P + kh, iw = aow[i] * S - P + kw;
                    if (ih >= 0 && ih < H && iw >= 0 && iw < W)
                        v0 = x[((an[i] * Ci + ci) * H + ih) * W + iw];
                }
                int k1 = kc + kp + 1;
                if (k1 < Kdim) {
                    int ci = k1 / KK; int rem = k1 - ci * KK;
                    int kh = rem / KH; int kw = rem - kh * KH;
                    int ih = aoh[i] * S - P + kh, iw = aow[i] * S - P + kw;
                    if (ih >= 0 && ih < H && iw >= 0 && iw < W)
                        v1 = x[((an[i] * Ci + ci) * H + ih) * W + iw];
                }
            }
            int msub = row >> 4, lrow = row & 15;
            int lsel = lrow + (((kp >> 3) & 1) << 4);
            int g = kp >> 4, q = (kp & 7) >> 1;
            int h0 = ((g << 2) + q) << 1;
            *(v2h*)&Al[pbuf][msub][lsel][h0] = (v2h){(_Float16)v0, (_Float16)v1};
        }

        // ---- stage B tile (16B vector load per thread, frag layout in LDS)
        {
            uint4 d = {0u, 0u, 0u, 0u};
            if (bval) d = *(const uint4*)(bsrc + kc);
            int nsub = bcol >> 4, lcol = bcol & 15;
            int lb = lcol + ((bseg & 1) << 4);
            int h0 = (bseg >> 1) << 3;
            *(uint4*)&Bl[pbuf][nsub][lb][h0] = d;
            if (bval && kc + 32 < Kpad)        // gfx1250 global_prefetch path
                __builtin_prefetch(bsrc + kc + 32, 0, 0);
        }

        __syncthreads();   // publishes buf[pbuf]; protects buf[pbuf^1] reuse

        // ---- compute: one A fragment reused against two B fragments
        v16h af  = *(const v16h*)&Al[pbuf][mw][lane][0];
        v16h bf0 = *(const v16h*)&Bl[pbuf][nw * 2 + 0][lane][0];
        v16h bf1 = *(const v16h*)&Bl[pbuf][nw * 2 + 1][lane][0];
        acc0 = __builtin_amdgcn_wmma_f32_16x16x32_f16(false, af, false, bf0,
                                                      (short)0, acc0, false, false);
        acc1 = __builtin_amdgcn_wmma_f32_16x16x32_f16(false, af, false, bf1,
                                                      (short)0, acc1, false, false);
    }

    // ---- epilogue: bias + activation (+ PixelShuffle r=2) store
#pragma unroll
    for (int half = 0; half < 2; ++half) {
        v8f acc = half ? acc1 : acc0;
        int ncol = nblock + nw * 32 + half * 16 + (lane & 15);
        if (ncol >= Co) continue;
        float bv = bias[ncol];
#pragma unroll
        for (int r = 0; r < 8; ++r) {
            int m = mblock + mw * 16 + ((lane >> 4) ? (r + 8) : r);
            if (m >= M) continue;
            float v = acc[r] + bv;
            if (act == 1)      v = (v > 0.f) ? v : (expf(v) - 1.f);   // CELU
            else if (act == 2) v = 1.f / (1.f + expf(-v));            // sigmoid
            int nn = m / OHW; int rr = m - nn * OHW;
            int o_h = rr / OW, o_w = rr - o_h * OW;
            if (ps) {   // in-ch c*4 + r1*2 + r2 -> (c, 2h+r1, 2w+r2)
                int Cp = Co >> 2, cc = ncol >> 2;
                int r1 = (ncol >> 1) & 1, r2 = ncol & 1;
                y[((size_t)(nn * Cp + cc) * (OH * 2) + (o_h * 2 + r1)) * (OW * 2)
                  + (o_w * 2 + r2)] = v;
            } else {
                y[((size_t)(nn * Co + ncol) * OH + o_h) * OW + o_w] = v;
            }
        }
    }
}

// ---------------------------------------------------------------------------
// GroupNorm: stats per (n, group) then elementwise apply (group is contiguous)
// ---------------------------------------------------------------------------
__global__ void gn_stats_kernel(const float* __restrict__ y, float* __restrict__ st,
                                int C, int HW, int G) {
    int n = blockIdx.x / G, g = blockIdx.x - n * G;
    int cs = C / G, cnt = cs * HW;
    const float* p = y + (size_t)(n * C + g * cs) * HW;
    float s = 0.f, s2 = 0.f;
    for (int i = threadIdx.x; i < cnt; i += 256) { float v = p[i]; s += v; s2 += v * v; }
    __shared__ float rs[256], rq[256];
    rs[threadIdx.x] = s; rq[threadIdx.x] = s2; __syncthreads();
    for (int k = 128; k > 0; k >>= 1) {
        if (threadIdx.x < k) { rs[threadIdx.x] += rs[threadIdx.x + k];
                               rq[threadIdx.x] += rq[threadIdx.x + k]; }
        __syncthreads();
    }
    if (threadIdx.x == 0) {
        float m   = rs[0] / (float)cnt;
        float var = rq[0] / (float)cnt - m * m;
        st[2 * blockIdx.x]     = m;
        st[2 * blockIdx.x + 1] = rsqrtf(var + 1e-5f);
    }
}

__global__ void gn_apply_kernel(float* __restrict__ y, const float* __restrict__ st,
                                const float* __restrict__ gamma, const float* __restrict__ beta,
                                int C, int HW, int G, int total) {
    int i = blockIdx.x * 256 + threadIdx.x;
    if (i >= total) return;
    int c = (i / HW) % C;
    int n = i / (C * HW);
    int g = c / (C / G);
    float m = st[2 * (n * G + g)], inv = st[2 * (n * G + g) + 1];
    y[i] = (y[i] - m) * inv * gamma[c] + beta[c];
}

// ---------------------------------------------------------------------------
// Scan: init, MLP+VQ (one block, B=16), posterior LSTMCell (prior LSTM is dead)
// ---------------------------------------------------------------------------
__global__ void init_state_kernel(const float* __restrict__ h0, const float* __restrict__ c0,
                                  float* __restrict__ hq, float* __restrict__ cq) {
    int i = blockIdx.x * 256 + threadIdx.x;
    if (i < 4096) { hq[i] = h0[i & 255]; cq[i] = c0[i & 255]; }
}

__global__ void step_mlp_vq_kernel(const float* __restrict__ encb, const float* __restrict__ cb,
                                   const float* __restrict__ w0, const float* __restrict__ b0,
                                   const float* __restrict__ w1, const float* __restrict__ b1,
                                   const float* __restrict__ w2, const float* __restrict__ b2,
                                   float* __restrict__ hq, float* __restrict__ zst,
                                   float* __restrict__ zout, float* __restrict__ klout, int t) {
    __shared__ float zc[16][384];
    __shared__ float z1s[16][128];
    __shared__ float z2s[16][128];
    __shared__ float zes[16][128];
    __shared__ float rd[256];
    __shared__ int   ri[256];
    __shared__ int   bidx[16];
    int tid = threadIdx.x;
    for (int i = tid; i < 16 * 256; i += 256) zc[i >> 8][i & 255] = hq[i];
    for (int i = tid; i < 16 * 128; i += 256) {
        int b = i >> 7, j = i & 127;
        zc[b][256 + j] = encb[(b * 32 + t) * 128 + j];
    }
    __syncthreads();
    for (int i = tid; i < 2048; i += 256) {          // layer 0: 384 -> 128, CELU
        int b = i >> 7, j = i & 127;
        float s = b0[j]; const float* wr = w0 + j * 384;
        for (int k = 0; k < 384; ++k) s += zc[b][k] * wr[k];
        z1s[b][j] = (s > 0.f) ? s : (expf(s) - 1.f);
    }
    __syncthreads();
    for (int i = tid; i < 2048; i += 256) {          // layer 1: 128 -> 128, CELU
        int b = i >> 7, j = i & 127;
        float s = b1[j]; const float* wr = w1 + j * 128;
        for (int k = 0; k < 128; ++k) s += z1s[b][k] * wr[k];
        z2s[b][j] = (s > 0.f) ? s : (expf(s) - 1.f);
    }
    __syncthreads();
    for (int i = tid; i < 2048; i += 256) {          // layer 2: 128 -> D
        int b = i >> 7, j = i & 127;
        float s = b2[j]; const float* wr = w2 + j * 128;
        for (int k = 0; k < 128; ++k) s += z2s[b][k] * wr[k];
        zes[b][j] = s;
    }
    __syncthreads();
    // VQ nearest neighbor: 16 threads per batch row, 32 codes each
    int b = tid >> 4, l = tid & 15;
    float bd = 3.4e38f; int bi = 0;
    for (int ii = 0; ii < 32; ++ii) {
        int c = l + (ii << 4);
        const float* cr = cb + c * 128;
        float d2 = 0.f;
        for (int d = 0; d < 128; ++d) { float df = zes[b][d] - cr[d]; d2 += df * df; }
        if (d2 < bd || (d2 == bd && c < bi)) { bd = d2; bi = c; }
    }
    rd[tid] = bd; ri[tid] = bi; __syncthreads();
    for (int s = 8; s > 0; s >>= 1) {
        if (l < s) {
            float od = rd[tid + s]; int oi = ri[tid + s];
            if (od < rd[tid] || (od == rd[tid] && oi < ri[tid])) { rd[tid] = od; ri[tid] = oi; }
        }
        __syncthreads();
    }
    if (l == 0) bidx[b] = ri[tid];
    __syncthreads();
    int cidx = bidx[b];
    const float* cr = cb + cidx * 128;
    float part = 0.f;
    for (int jj = 0; jj < 8; ++jj) {
        int d = l + (jj << 4);
        float zev = zes[b][d], zqv = cr[d];
        float zs  = zev + (zqv - zev);               // straight-through forward
        zst[b * 128 + d]             = zs;
        zout[(b * 32 + t) * 128 + d] = zs;
        float df = zev - zqv;
        part += df * df + df * df;                   // codebook + beta*commit (beta=1)
    }
    rd[tid] = part; __syncthreads();
    for (int s = 8; s > 0; s >>= 1) { if (l < s) rd[tid] += rd[tid + s]; __syncthreads(); }
    if (l == 0) klout[b * 32 + t] = rd[tid] * (1.f / 128.f);
}

__global__ void step_lstm_kernel(const float* __restrict__ zst, const float* __restrict__ wih,
                                 const float* __restrict__ whh, const float* __restrict__ bih,
                                 const float* __restrict__ bhh,
                                 float* __restrict__ hq, float* __restrict__ cq) {
    __shared__ float g[1024];
    __shared__ float xv[128];
    __shared__ float hv[256];
    int b = blockIdx.x, tid = threadIdx.x;
    if (tid < 128) xv[tid] = zst[b * 128 + tid];
    hv[tid] = hq[b * 256 + (tid & 255)];
    __syncthreads();
    for (int j = tid; j < 1024; j += 256) {
        float s = bih[j] + bhh[j];
        const float* wi = wih + j * 128;
        const float* wh = whh + j * 256;
        for (int k = 0; k < 128; ++k) s += xv[k] * wi[k];
        for (int k = 0; k < 256; ++k) s += hv[k] * wh[k];
        g[j] = s;
    }
    __syncthreads();
    int j = tid;  // gate order i,f,g,o
    float iv = g[j], fv = g[j + 256], gv = g[j + 512], ov = g[j + 768];
    float si = 1.f / (1.f + expf(-iv));
    float sf = 1.f / (1.f + expf(-fv));
    float so = 1.f / (1.f + expf(-ov));
    float c  = sf * cq[b * 256 + j] + si * tanhf(gv);
    cq[b * 256 + j] = c;
    hq[b * 256 + j] = so * tanhf(c);
}

// ---------------------------------------------------------------------------
// Host-side launch helpers
// ---------------------------------------------------------------------------
static void launch_conv(const float* x, const float* w, const float* bias, float* y,
                        _Float16* w16, int N, int Ci, int H, int W, int Co,
                        int KH, int S, int P, int act, int ps, hipStream_t stream) {
    int OH   = (H + 2 * P - KH) / S + 1;
    int OW   = OH;
    int Kdim = Ci * KH * KH;
    int Kpad = (Kdim + 31) & ~31;
    int tot  = Co * Kpad;
    wcvt_kernel<<<(tot + 255) / 256, 256, 0, stream>>>(w, w16, Co, Kdim, Kpad);
    int M = N * OH * OW;
    dim3 grid((M + 63) / 64, (Co + 63) / 64);
    conv_wmma_kernel<<<grid, 256, 0, stream>>>(x, w16, bias, y, M, Co, Ci, H, W,
                                               KH, S, P, OH, OW, Kdim, Kpad, act, ps);
}

static void launch_gn(float* y, float* st, const float* gamma, const float* beta,
                      int N, int C, int HW, int G, hipStream_t stream) {
    gn_stats_kernel<<<N * G, 256, 0, stream>>>(y, st, C, HW, G);
    int total = N * C * HW;
    gn_apply_kernel<<<(total + 255) / 256, 256, 0, stream>>>(y, st, gamma, beta, C, HW, G, total);
}

extern "C" void kernel_launch(void* const* d_in, const int* in_sizes, int n_in,
                              void* d_out, int out_size, void* d_ws, size_t ws_size,
                              hipStream_t stream) {
    (void)in_sizes; (void)n_in; (void)out_size; (void)ws_size;
    auto F = [&](int i) { return (const float*)d_in[i]; };
    // Input order: 0 seq; 1-4 enc_w; 5-8 enc_b; 9-12 enc_gng; 13-16 enc_gnb;
    // 17 enc_fc_w; 18 enc_fc_b; 19 codebook; 20 dec_fc_w; 21 dec_fc_b;
    // 22-25 dec_w; 26-29 dec_b; 30-32 dec_gng; 33-35 dec_gnb;
    // 36-41 post (wih,whh,bih,bhh,h0,c0); 42-47 prior (dead); 48-50 mlp_w; 51-53 mlp_b.

    float* base = (float*)d_ws;
    float* actA = base;                         // 33,554,432 f32 (ping)
    float* actB = actA + 33554432;              // 33,554,432 f32 (pong)
    float* encb = actB + 33554432;              // 65,536  (512 x 128 encoder output)
    float* hq   = encb + 65536;                 // 16 x 256
    float* cq   = hq + 4096;                    // 16 x 256
    float* zst  = cq + 4096;                    // 16 x 128
    float* gst  = zst + 8192;                   // GN stats (<= 32,768)
    _Float16* w16 = (_Float16*)(gst + 32768);   // staged f16 weights (<= 2M halves)

    float* out_bg = (float*)d_out;              // 512 x 3 x 64 x 64
    float* out_z  = out_bg + 512 * 3 * 64 * 64; // 512 x 128 (B,T,D)
    float* out_kl = out_z + 512 * 128;          // 512 (B,T)

    // ---------------- encoder: 4x (conv -> CELU -> GroupNorm) ----------------
    launch_conv(F(0), F(1), F(5), actA, w16, 512,   3, 64, 64,  64, 7, 2, 3, 1, 0, stream);
    launch_gn(actA, gst, F(9),  F(13), 512,  64, 32 * 32,  4, stream);
    launch_conv(actA, F(2), F(6), actB, w16, 512,  64, 32, 32, 128, 3, 2, 1, 1, 0, stream);
    launch_gn(actB, gst, F(10), F(14), 512, 128, 16 * 16,  8, stream);
    launch_conv(actB, F(3), F(7), actA, w16, 512, 128, 16, 16, 256, 3, 2, 1, 1, 0, stream);
    launch_gn(actA, gst, F(11), F(15), 512, 256,  8 * 8,  16, stream);
    launch_conv(actA, F(4), F(8), actB, w16, 512, 256,  8,  8, 512, 3, 2, 1, 1, 0, stream);
    launch_gn(actB, gst, F(12), F(16), 512, 512,  4 * 4,  32, stream);
    // encoder FC as 1x1 conv: (512, 8192) -> (512, 128)
    launch_conv(actB, F(17), F(18), encb, w16, 512, 8192, 1, 1, 128, 1, 1, 0, 0, 0, stream);

    // ---------------- sequential scan (MLP -> VQ -> posterior LSTM) ----------
    init_state_kernel<<<16, 256, 0, stream>>>(F(40), F(41), hq, cq);
    for (int t = 0; t < 32; ++t) {
        step_mlp_vq_kernel<<<1, 256, 0, stream>>>(encb, F(19), F(48), F(51), F(49), F(52),
                                                  F(50), F(53), hq, zst, out_z, out_kl, t);
        step_lstm_kernel<<<16, 256, 0, stream>>>(zst, F(36), F(37), F(38), F(39), hq, cq);
    }

    // ---------------- decoder: FC, 3x (conv->PS->CELU->GN), conv->PS->sigmoid
    launch_conv(out_z, F(20), F(21), actA, w16, 512, 128, 1, 1, 2048, 1, 1, 0, 0, 0, stream);
    launch_conv(actA, F(22), F(26), actB, w16, 512, 128,  4,  4, 256, 3, 1, 1, 1, 1, stream);
    launch_gn(actB, gst, F(30), F(33), 512, 64,  8 * 8,  4, stream);
    launch_conv(actB, F(23), F(27), actA, w16, 512,  64,  8,  8, 128, 3, 1, 1, 1, 1, stream);
    launch_gn(actA, gst, F(31), F(34), 512, 32, 16 * 16, 2, stream);
    launch_conv(actA, F(24), F(28), actB, w16, 512,  32, 16, 16,  64, 3, 1, 1, 1, 1, stream);
    launch_gn(actB, gst, F(32), F(35), 512, 16, 32 * 32, 1, stream);
    launch_conv(actB, F(25), F(29), out_bg, w16, 512, 16, 32, 32, 12, 3, 1, 1, 2, 1, stream);
}
